// DTAModel_17411797418187
// MI455X (gfx1250) — matrix operand order repaired and checked by reading, not compiled, not run on hardware
//
#include <hip/hip_runtime.h>
#include <hip/hip_bf16.h>

#define N_NODES 32768
#define N_EDGES 131072
#define BATCHSZ 512
#define SEQ_LEN 1000
#define VOCABSZ 26
#define EMB_DIM 128

typedef __attribute__((ext_vector_type(16))) _Float16 v16h;
typedef __attribute__((ext_vector_type(8)))  _Float16 v8h;   // 16 bytes
typedef __attribute__((ext_vector_type(8)))  float    v8f;

static inline int cdiv(int a, int b) { return (a + b - 1) / b; }

// ---------------------------------------------------------------------------
// Direct-from-memory WMMA fragment loaders (16-bit, same contiguous k order in A and B, wave32).
// A matrix row-major [M][K]; fragment rows m0..m0+15, K-chunk k0..k0+31.
//   lane(m = lane&15, hi = lane>>4): halves j -> K=hi*16+j (32B contiguous)
// B supplied transposed, row-major [N][K]; fragment cols n0..n0+15.
//   lane(n = lane&15, khi = lane>>4): halves j -> K=khi*16+j (32B contiguous)
// ---------------------------------------------------------------------------
__device__ __forceinline__ v16h combine16(v8h lo, v8h hp) {
    v16h r;
#pragma unroll
    for (int j = 0; j < 8; ++j) { r[j] = lo[j]; r[8 + j] = hp[j]; }
    return r;
}

template <int K>
__device__ __forceinline__ v16h load_frag_A(const _Float16* __restrict__ base /* A+m0*K+k0 */) {
    int lane = threadIdx.x & 31;
    int m = lane & 15, hi = lane >> 4;
    const _Float16* row = base + (size_t)m * K;
    v8h lo = *(const v8h*)(row + hi * 16);
    v8h hp = *(const v8h*)(row + hi * 16 + 8);
    return combine16(lo, hp);
}

template <int K>
__device__ __forceinline__ v16h load_frag_B(const _Float16* __restrict__ base /* Bt+n0*K+k0 */) {
    int lane = threadIdx.x & 31;
    int n = lane & 15, khi = lane >> 4;
    const v8h* p = (const v8h*)(base + (size_t)n * K + khi * 16);
    return combine16(p[0], p[1]);
}

// ---------------------------------------------------------------------------
// Register-tiled WMMA GEMM: C[M,N] = act(A[M,K] @ Bt[N,K]^T + bias[N])
// One wave -> 16(M) x 64(N) via 4 accumulators. No LDS, no barriers.
// K % 32 == 0 (template const), M % 16 == 0, N % 64 == 0.
// ---------------------------------------------------------------------------
template <int K, bool RELU, bool OUT_F16>
__global__ void gemm_wmma_kernel(const _Float16* __restrict__ A,
                                 const _Float16* __restrict__ Bt,
                                 const float* __restrict__ bias,
                                 void* __restrict__ Cout,
                                 int M, int N) {
    int wave = threadIdx.x >> 5;
    int lane = threadIdx.x & 31;
    int tiles_n = N >> 6;                       // groups of 64 cols
    int tile = blockIdx.x * 8 + wave;
    int total = (M >> 4) * tiles_n;
    if (tile >= total) tile = total - 1;        // uniform clamp (dup tile)
    int m0 = (tile / tiles_n) * 16;
    int n0 = (tile % tiles_n) * 64;
    v8f acc[4] = {};
    for (int k0 = 0; k0 < K; k0 += 32) {
        v16h a = load_frag_A<K>(A + (size_t)m0 * K + k0);
#pragma unroll
        for (int t = 0; t < 4; ++t) {
            v16h b = load_frag_B<K>(Bt + (size_t)(n0 + t * 16) * K + k0);
            acc[t] = __builtin_amdgcn_wmma_f32_16x16x32_f16(false, a, false, b,
                                                            (short)0, acc[t], false, false);
        }
    }
    int n = lane & 15, hi = lane >> 4;
#pragma unroll
    for (int t = 0; t < 4; ++t) {
        int nc = n0 + t * 16 + n;
        float bv = bias ? bias[nc] : 0.0f;
#pragma unroll
        for (int v = 0; v < 8; ++v) {
            int m = m0 + v + hi * 8;
            float val = acc[t][v] + bv;
            if (RELU) val = fmaxf(val, 0.0f);
            if (OUT_F16) ((_Float16*)Cout)[(size_t)m * N + nc] = (_Float16)val;
            else         ((float*)Cout)[(size_t)m * N + nc] = val;
        }
    }
}

// ---------------------------------------------------------------------------
// Conv1d as implicit GEMM via WMMA (direct loads, register-tiled 16co x 64l).
// in    : f16 [B][Lin][CI] channel-fastest
// wflat : f16 [co][KS*CI] with Kidx = k*CI + c
// out   : f16 [B][Lout][co], bias + relu fused.
// CI in {32,64,128} (power of two), co % 16 == 0.
// ---------------------------------------------------------------------------
template <int CI, int KS>
__global__ void conv1d_wmma_kernel(const _Float16* __restrict__ in,
                                   const _Float16* __restrict__ wflat,
                                   const float* __restrict__ bias,
                                   _Float16* __restrict__ out,
                                   int Lin, int Lout, int co, int pad) {
    constexpr int K = CI * KS;
    int b = blockIdx.x;
    int wave = threadIdx.x >> 5;
    int lane = threadIdx.x & 31;
    int lt4 = (Lout + 63) >> 6;                 // groups of 64 positions
    int tile = blockIdx.y * 8 + wave;
    int total = (co >> 4) * lt4;
    if (tile >= total) tile = total - 1;        // uniform clamp
    int m0 = (tile / lt4) * 16;                 // output-channel offset
    int n0 = (tile % lt4) * 64;                 // position offset
    const _Float16* inb = in + (size_t)b * Lin * CI;
    int n = lane & 15, khi = lane >> 4;
    v8f acc[4] = {};
    for (int k0 = 0; k0 < K; k0 += 32) {
        int ktap = k0 / CI;                     // constant-divisor shift
        int c0   = k0 % CI;
        v16h a = load_frag_A<K>(wflat + (size_t)m0 * K + k0);
#pragma unroll
        for (int t = 0; t < 4; ++t) {
            int p = n0 + t * 16 + n + ktap - pad;   // input position (per lane)
            v8h z = {};
            v8h lo = z, hp = z;
            if (p >= 0 && p < Lin) {
                const v8h* q = (const v8h*)(inb + (size_t)p * CI + c0 + khi * 16);
                lo = q[0]; hp = q[1];
            }
            v16h bf = combine16(lo, hp);
            acc[t] = __builtin_amdgcn_wmma_f32_16x16x32_f16(false, a, false, bf,
                                                            (short)0, acc[t], false, false);
        }
    }
    int hi = lane >> 4;
#pragma unroll
    for (int t = 0; t < 4; ++t) {
        int l = n0 + t * 16 + n;
        if (l < Lout) {
#pragma unroll
            for (int v = 0; v < 8; ++v) {
                int m = m0 + v + hi * 8;
                float val = fmaxf(acc[t][v] + bias[m], 0.0f);
                out[((size_t)b * Lout + l) * co + m] = (_Float16)val;
            }
        }
    }
}

// ---------------------------------------------------------------------------
// Small helper kernels
// ---------------------------------------------------------------------------
__global__ void fill_f32_kernel(float* p, float v, size_t n) {
    size_t i = (size_t)blockIdx.x * blockDim.x + threadIdx.x;
    if (i < n) p[i] = v;
}

__global__ void deg_accum_kernel(const int* __restrict__ dst, float* deg, int ne) {
    int e = blockIdx.x * blockDim.x + threadIdx.x;
    if (e < ne) atomicAdd(&deg[dst[e]], 1.0f);
}

__global__ void make_dis_kernel(const float* __restrict__ deg, float* dis, int n) {
    int i = blockIdx.x * blockDim.x + threadIdx.x;
    if (i < n) dis[i] = rsqrtf(deg[i]);
}

// xw1 = x[32768,5] @ W1[5,64]  (K=5 tiny -> scalar)
__global__ void xw1_kernel(const float* __restrict__ x, const float* __restrict__ W1,
                           float* __restrict__ xw) {
    int idx = blockIdx.x * blockDim.x + threadIdx.x;
    int node = idx >> 6, j = idx & 63;
    if (node < N_NODES) {
        float s = 0.0f;
#pragma unroll
        for (int k = 0; k < 5; ++k) s += x[node * 5 + k] * W1[k * 64 + j];
        xw[(size_t)node * 64 + j] = s;
    }
}

// agg[v][f] = xw[v][f]*dis[v]^2 + bias[f]    (self-loop term + bias)
__global__ void mp_init_kernel(const float* __restrict__ xw, const float* __restrict__ dis,
                               const float* __restrict__ bias, float* __restrict__ agg,
                               int F) {
    size_t idx = (size_t)blockIdx.x * blockDim.x + threadIdx.x;
    if (idx < (size_t)N_NODES * F) {
        int v = (int)(idx / F), f = (int)(idx % F);
        float d = dis[v];
        agg[idx] = xw[idx] * d * d + bias[f];
    }
}

// agg[dst][f] += xw[src][f] * dis[src] * dis[dst]
__global__ void mp_edges_kernel(const int* __restrict__ src, const int* __restrict__ dst,
                                const float* __restrict__ xw, const float* __restrict__ dis,
                                float* __restrict__ agg, int F) {
    size_t idx = (size_t)blockIdx.x * blockDim.x + threadIdx.x;
    if (idx < (size_t)N_EDGES * F) {
        int e = (int)(idx / F), f = (int)(idx % F);
        int s = src[e], d = dst[e];
        atomicAdd(&agg[(size_t)d * F + f], xw[(size_t)s * F + f] * dis[s] * dis[d]);
    }
}

// Per-column mean/var (ddof=0). One block per column.
__global__ void bn_stats_kernel(const float* __restrict__ a, float* mean, float* var, int F) {
    int f = blockIdx.x;
    float s = 0.0f, s2 = 0.0f;
    for (int i = threadIdx.x; i < N_NODES; i += blockDim.x) {
        float v = a[(size_t)i * F + f];
        s += v; s2 += v * v;
    }
    __shared__ float bs[256], bq[256];
    bs[threadIdx.x] = s; bq[threadIdx.x] = s2;
    __syncthreads();
    for (int st = 128; st > 0; st >>= 1) {
        if (threadIdx.x < st) { bs[threadIdx.x] += bs[threadIdx.x + st];
                                bq[threadIdx.x] += bq[threadIdx.x + st]; }
        __syncthreads();
    }
    if (threadIdx.x == 0) {
        float m = bs[0] / (float)N_NODES;
        mean[f] = m;
        var[f] = bq[0] / (float)N_NODES - m * m;
    }
}

// h = f16(relu(g*(a-mean)*rsqrt(var+eps)+bt))
__global__ void bn_relu_f16_kernel(const float* __restrict__ a, const float* __restrict__ mean,
                                   const float* __restrict__ var, const float* __restrict__ g,
                                   const float* __restrict__ bt, _Float16* __restrict__ h,
                                   int F) {
    size_t idx = (size_t)blockIdx.x * blockDim.x + threadIdx.x;
    if (idx < (size_t)N_NODES * F) {
        int f = (int)(idx % F);
        float y = g[f] * (a[idx] - mean[f]) * rsqrtf(var[f] + 1e-5f) + bt[f];
        h[idx] = (_Float16)fmaxf(y, 0.0f);
    }
}

// W[K][N] fp32 -> Wt[N][K] f16
__global__ void transpose_w_kernel(const float* __restrict__ W, _Float16* __restrict__ Wt,
                                   int K, int N) {
    int idx = blockIdx.x * blockDim.x + threadIdx.x;
    if (idx < K * N) {
        int k = idx / N, n = idx % N;
        Wt[(size_t)n * K + k] = (_Float16)W[idx];
    }
}

// conv weights [co][ci][ks] fp32 -> wflat [co][ks*ci] f16 with Kidx = k*ci + c
__global__ void repack_conv_w_kernel(const float* __restrict__ w, _Float16* __restrict__ wf,
                                     int co, int ci, int ks) {
    int idx = blockIdx.x * blockDim.x + threadIdx.x;
    if (idx < co * ci * ks) {
        int o = idx / (ci * ks);
        int c = (idx / ks) % ci;
        int k = idx % ks;
        wf[(size_t)o * (ks * ci) + k * ci + c] = (_Float16)w[idx];
    }
}

__global__ void f32_to_f16_kernel(const float* __restrict__ a, _Float16* __restrict__ b, int n) {
    int i = blockIdx.x * blockDim.x + threadIdx.x;
    if (i < n) b[i] = (_Float16)a[i];
}

// act0[b][l][c] = emb16[seq[b*L+l]][c]
__global__ void embed_gather_kernel(const int* __restrict__ seq,
                                    const _Float16* __restrict__ emb16,
                                    _Float16* __restrict__ out) {
    size_t idx = (size_t)blockIdx.x * blockDim.x + threadIdx.x;
    if (idx < (size_t)BATCHSZ * SEQ_LEN * EMB_DIM) {
        size_t bl = idx / EMB_DIM;
        int c = (int)(idx % EMB_DIM);
        out[idx] = emb16[(size_t)seq[bl] * EMB_DIM + c];
    }
}

// drug pooling: atomic sums (batch sorted but arbitrary counts)
__global__ void pool_accum_kernel(const _Float16* __restrict__ h, const int* __restrict__ batch,
                                  float* __restrict__ pool, float* __restrict__ cnt, int F) {
    size_t idx = (size_t)blockIdx.x * blockDim.x + threadIdx.x;
    if (idx < (size_t)N_NODES * F) {
        int v = (int)(idx / F), f = (int)(idx % F);
        int bb = batch[v];
        atomicAdd(&pool[(size_t)bb * F + f], (float)h[idx]);
        if (f == 0) atomicAdd(&cnt[bb], 1.0f);
    }
}

__global__ void pool_finish_kernel(const float* __restrict__ pool, const float* __restrict__ cnt,
                                   _Float16* __restrict__ cat, int F, int ldcat) {
    int idx = blockIdx.x * blockDim.x + threadIdx.x;
    if (idx < BATCHSZ * F) {
        int b = idx / F, f = idx % F;
        cat[(size_t)b * ldcat + f] = (_Float16)(pool[idx] / fmaxf(cnt[b], 1.0f));
    }
}

// max over L of c3[b][l][m], write into concat cols [off, off+co)
__global__ void maxpool_kernel(const _Float16* __restrict__ c3, _Float16* __restrict__ cat,
                               int L, int co, int ldcat, int off) {
    int idx = blockIdx.x * blockDim.x + threadIdx.x;
    if (idx < BATCHSZ * co) {
        int b = idx / co, m = idx % co;
        float mx = -3.0e38f;
        const _Float16* p = c3 + (size_t)b * L * co + m;
        for (int l = 0; l < L; ++l) mx = fmaxf(mx, (float)p[(size_t)l * co]);
        cat[(size_t)b * ldcat + off + m] = (_Float16)mx;
    }
}

// out[b] = sum_k c[b][k]*w[k] + b3
__global__ void fc3_kernel(const _Float16* __restrict__ c, const float* __restrict__ w,
                           const float* __restrict__ b3, float* __restrict__ out, int K) {
    int b = blockIdx.x * blockDim.x + threadIdx.x;
    if (b < BATCHSZ) {
        float s = b3[0];
        for (int k = 0; k < K; ++k) s += (float)c[(size_t)b * K + k] * w[k];
        out[b] = s;
    }
}

// ---------------------------------------------------------------------------
extern "C" void kernel_launch(void* const* d_in, const int* in_sizes, int n_in,
                              void* d_out, int out_size, void* d_ws, size_t ws_size,
                              hipStream_t stream) {
    (void)in_sizes; (void)n_in; (void)out_size; (void)ws_size;
    const float* x    = (const float*)d_in[0];
    const int*   eidx = (const int*)d_in[1];
    const int*   batch= (const int*)d_in[2];
    const int*   seq  = (const int*)d_in[3];
    const float *W1=(const float*)d_in[4],  *b1=(const float*)d_in[5];
    const float *g1=(const float*)d_in[6],  *bt1=(const float*)d_in[7];
    const float *W2=(const float*)d_in[8],  *b2=(const float*)d_in[9];
    const float *g2=(const float*)d_in[10], *bt2=(const float*)d_in[11];
    const float *W3=(const float*)d_in[12], *b3=(const float*)d_in[13];
    const float *g3=(const float*)d_in[14], *bt3=(const float*)d_in[15];
    const float *emb=(const float*)d_in[16];
    const float *ck1=(const float*)d_in[17], *cb1=(const float*)d_in[18];
    const float *ck2=(const float*)d_in[19], *cb2=(const float*)d_in[20];
    const float *ck3=(const float*)d_in[21], *cb3=(const float*)d_in[22];
    const float *fw1=(const float*)d_in[23], *fb1=(const float*)d_in[24];
    const float *fw2=(const float*)d_in[25], *fb2=(const float*)d_in[26];
    const float *fw3=(const float*)d_in[27], *fb3=(const float*)d_in[28];
    const int* src = eidx;
    const int* dst = eidx + N_EDGES;
    float* out = (float*)d_out;

    // ---- carve workspace (256B aligned) ----
    char* wsp = (char*)d_ws;
    size_t off = 0;
    auto carve = [&](size_t bytes) -> void* {
        void* p = wsp + off;
        off = (off + bytes + 255) & ~(size_t)255;
        return p;
    };
    _Float16* act0 = (_Float16*)carve((size_t)BATCHSZ * SEQ_LEN * EMB_DIM * 2); // 131 MB
    _Float16* c1   = (_Float16*)carve((size_t)BATCHSZ * 999 * 32 * 2);          // 33 MB
    _Float16* c3   = (_Float16*)carve((size_t)BATCHSZ * 997 * 96 * 2);          // 98 MB
    _Float16* c2   = act0;                                                      // 65 MB reuse
    float* xw  = (float*)carve((size_t)N_NODES * 128 * 4);
    float* agg = (float*)carve((size_t)N_NODES * 128 * 4);
    _Float16* hA = (_Float16*)carve((size_t)N_NODES * 128 * 2);
    _Float16* hB = (_Float16*)carve((size_t)N_NODES * 128 * 2);
    float* deg  = (float*)carve(N_NODES * 4);
    float* dis  = (float*)carve(N_NODES * 4);
    float* mean = (float*)carve(128 * 4);
    float* var  = (float*)carve(128 * 4);
    float* pool = (float*)carve((size_t)BATCHSZ * 128 * 4);
    float* cnt  = (float*)carve(BATCHSZ * 4);
    _Float16* w2t  = (_Float16*)carve(128 * 64 * 2);
    _Float16* w3t  = (_Float16*)carve(128 * 128 * 2);
    _Float16* fw1t = (_Float16*)carve(512 * 224 * 2);
    _Float16* fw2t = (_Float16*)carve(256 * 512 * 2);
    _Float16* cw1  = (_Float16*)carve(32 * 512 * 2);
    _Float16* cw2  = (_Float16*)carve(64 * 192 * 2);
    _Float16* cw3  = (_Float16*)carve(96 * 512 * 2);
    _Float16* emb16= (_Float16*)carve(VOCABSZ * EMB_DIM * 2);
    _Float16* cat  = (_Float16*)carve((size_t)BATCHSZ * 224 * 2);
    _Float16* fc1o = (_Float16*)carve((size_t)BATCHSZ * 512 * 2);
    _Float16* fc2o = (_Float16*)carve((size_t)BATCHSZ * 256 * 2);

    const int B256 = 256;

    // ---- weight prep (f16 conversion / transpose / repack) ----
    transpose_w_kernel<<<cdiv(64 * 128, B256), B256, 0, stream>>>(W2, w2t, 64, 128);
    transpose_w_kernel<<<cdiv(128 * 128, B256), B256, 0, stream>>>(W3, w3t, 128, 128);
    transpose_w_kernel<<<cdiv(224 * 512, B256), B256, 0, stream>>>(fw1, fw1t, 224, 512);
    transpose_w_kernel<<<cdiv(512 * 256, B256), B256, 0, stream>>>(fw2, fw2t, 512, 256);
    repack_conv_w_kernel<<<cdiv(32 * 128 * 4, B256), B256, 0, stream>>>(ck1, cw1, 32, 128, 4);
    repack_conv_w_kernel<<<cdiv(64 * 32 * 6, B256), B256, 0, stream>>>(ck2, cw2, 64, 32, 6);
    repack_conv_w_kernel<<<cdiv(96 * 64 * 8, B256), B256, 0, stream>>>(ck3, cw3, 96, 64, 8);
    f32_to_f16_kernel<<<cdiv(VOCABSZ * EMB_DIM, B256), B256, 0, stream>>>(emb, emb16, VOCABSZ * EMB_DIM);

    // ---- GCN degree normalization ----
    fill_f32_kernel<<<cdiv(N_NODES, B256), B256, 0, stream>>>(deg, 1.0f, N_NODES);
    deg_accum_kernel<<<cdiv(N_EDGES, B256), B256, 0, stream>>>(dst, deg, N_EDGES);
    make_dis_kernel<<<cdiv(N_NODES, B256), B256, 0, stream>>>(deg, dis, N_NODES);

    // ---- GCN layer 1 (F=64) ----
    xw1_kernel<<<cdiv(N_NODES * 64, B256), B256, 0, stream>>>(x, W1, xw);
    mp_init_kernel<<<cdiv(N_NODES * 64, B256), B256, 0, stream>>>(xw, dis, b1, agg, 64);
    mp_edges_kernel<<<cdiv(N_EDGES * 64, B256), B256, 0, stream>>>(src, dst, xw, dis, agg, 64);
    bn_stats_kernel<<<64, B256, 0, stream>>>(agg, mean, var, 64);
    bn_relu_f16_kernel<<<cdiv(N_NODES * 64, B256), B256, 0, stream>>>(agg, mean, var, g1, bt1, hA, 64);

    // ---- GCN layer 2: xw = hA[32768,64] @ W2 -> [32768,128] (WMMA) ----
    {
        int total = (N_NODES / 16) * (128 / 64);
        gemm_wmma_kernel<64, false, false><<<cdiv(total, 8), B256, 0, stream>>>(
            hA, w2t, nullptr, xw, N_NODES, 128);
    }
    mp_init_kernel<<<cdiv(N_NODES * 128, B256), B256, 0, stream>>>(xw, dis, b2, agg, 128);
    mp_edges_kernel<<<cdiv(N_EDGES * 128, B256), B256, 0, stream>>>(src, dst, xw, dis, agg, 128);
    bn_stats_kernel<<<128, B256, 0, stream>>>(agg, mean, var, 128);
    bn_relu_f16_kernel<<<cdiv(N_NODES * 128, B256), B256, 0, stream>>>(agg, mean, var, g2, bt2, hB, 128);

    // ---- GCN layer 3 (WMMA) ----
    {
        int total = (N_NODES / 16) * (128 / 64);
        gemm_wmma_kernel<128, false, false><<<cdiv(total, 8), B256, 0, stream>>>(
            hB, w3t, nullptr, xw, N_NODES, 128);
    }
    mp_init_kernel<<<cdiv(N_NODES * 128, B256), B256, 0, stream>>>(xw, dis, b3, agg, 128);
    mp_edges_kernel<<<cdiv(N_EDGES * 128, B256), B256, 0, stream>>>(src, dst, xw, dis, agg, 128);
    bn_stats_kernel<<<128, B256, 0, stream>>>(agg, mean, var, 128);
    bn_relu_f16_kernel<<<cdiv(N_NODES * 128, B256), B256, 0, stream>>>(agg, mean, var, g3, bt3, hA, 128);

    // ---- drug mean pooling -> concat[:, 0:128] ----
    fill_f32_kernel<<<cdiv(BATCHSZ * 128, B256), B256, 0, stream>>>(pool, 0.0f, (size_t)BATCHSZ * 128);
    fill_f32_kernel<<<cdiv(BATCHSZ, B256), B256, 0, stream>>>(cnt, 0.0f, BATCHSZ);
    pool_accum_kernel<<<cdiv(N_NODES * 128, B256), B256, 0, stream>>>(hA, batch, pool, cnt, 128);
    pool_finish_kernel<<<cdiv(BATCHSZ * 128, B256), B256, 0, stream>>>(pool, cnt, cat, 128, 224);

    // ---- protein: embedding gather -> f16 [B][L][128] ----
    {
        size_t n = (size_t)BATCHSZ * SEQ_LEN * EMB_DIM;
        embed_gather_kernel<<<(unsigned)((n + B256 - 1) / B256), B256, 0, stream>>>(seq, emb16, act0);
    }
    // conv1: 128 -> 32, ks=4, pad=1, 1000 -> 999
    {
        int lt4 = cdiv(999, 64), total = (32 / 16) * lt4;
        dim3 grid(BATCHSZ, cdiv(total, 8));
        conv1d_wmma_kernel<128, 4><<<grid, B256, 0, stream>>>(act0, cw1, cb1, c1, 1000, 999, 32, 1);
    }
    // conv2: 32 -> 64, ks=6, pad=2, 999 -> 998   (out reuses act0 region)
    {
        int lt4 = cdiv(998, 64), total = (64 / 16) * lt4;
        dim3 grid(BATCHSZ, cdiv(total, 8));
        conv1d_wmma_kernel<32, 6><<<grid, B256, 0, stream>>>(c1, cw2, cb2, c2, 999, 998, 64, 2);
    }
    // conv3: 64 -> 96, ks=8, pad=3, 998 -> 997
    {
        int lt4 = cdiv(997, 64), total = (96 / 16) * lt4;
        dim3 grid(BATCHSZ, cdiv(total, 8));
        conv1d_wmma_kernel<64, 8><<<grid, B256, 0, stream>>>(c2, cw3, cb3, c3, 998, 997, 96, 3);
    }
    // global max pool -> concat[:, 128:224]
    maxpool_kernel<<<cdiv(BATCHSZ * 96, B256), B256, 0, stream>>>(c3, cat, 997, 96, 224, 128);

    // ---- regressor head (WMMA) ----
    {
        int total = (BATCHSZ / 16) * (512 / 64);
        gemm_wmma_kernel<224, true, true><<<cdiv(total, 8), B256, 0, stream>>>(
            cat, fw1t, fb1, fc1o, BATCHSZ, 512);
    }
    {
        int total = (BATCHSZ / 16) * (256 / 64);
        gemm_wmma_kernel<512, true, true><<<cdiv(total, 8), B256, 0, stream>>>(
            fc1o, fw2t, fb2, fc2o, BATCHSZ, 256);
    }
    fc3_kernel<<<cdiv(BATCHSZ, B256), B256, 0, stream>>>(fc2o, fw3, fb3, out, 256);
}
